// Graph_ConvNet_LeNet5_76914274337414
// MI455X (gfx1250) — matrix operand order repaired and checked by reading, not compiled, not run on hardware
//
#include <hip/hip_runtime.h>
#include <cstdint>
#include <cstddef>

// ---------------------------------------------------------------------------
// Graph ChebConv LeNet5 for MI455X (gfx1250, wave32, WMMA bf16 16x16x32)
// ---------------------------------------------------------------------------

typedef __attribute__((ext_vector_type(16))) __bf16 v16bf;
typedef __attribute__((ext_vector_type(8)))  float  v8f;

constexpr int   DV     = 4096;     // nodes, layer 1
constexpr int   BATCH  = 64;
constexpr int   V2     = 2048;     // nodes after pool1
constexpr int   V3     = 1024;     // nodes after pool2
constexpr int   F1     = 32;       // CL1_F
constexpr int   F2     = 64;       // CL2_F
constexpr int   KCH    = 25;       // Chebyshev order
constexpr int   NNZ1   = DV * 8;
constexpr int   NNZ2   = V2 * 8;
constexpr int   NC2    = F1 * BATCH;    // 2048: conv2 SpMM column width (col = b*32+fin)
constexpr int   FC1IN  = F2 * V3;       // 65536
constexpr int   FC1F   = 512;
constexpr int   FC2F   = 10;
constexpr int   W2ROW  = F1 * KCH;      // 800
constexpr int   KSPLIT = 32;            // FC1 K-dim split factor
constexpr int   KSLC   = FC1IN / KSPLIT; // 2048 K per slice

// ---- helpers ---------------------------------------------------------------

__device__ __forceinline__ unsigned short f2bf(float f) {
  unsigned int u = __float_as_uint(f);
  u += 0x7FFFu + ((u >> 16) & 1u);          // round-to-nearest-even
  return (unsigned short)(u >> 16);
}
__device__ __forceinline__ __bf16 us2bf(unsigned short u) {
  return __builtin_bit_cast(__bf16, u);
}

// ---- elementwise -----------------------------------------------------------

__global__ __launch_bounds__(256) void k_zero(float* __restrict__ p, int n) {
  int t = blockIdx.x * blockDim.x + threadIdx.x;
  if (t < n) p[t] = 0.0f;
}

// out = as*a + bs*b
__global__ __launch_bounds__(256) void k_axpby(float* __restrict__ out, float as,
                                               const float* __restrict__ a, float bs,
                                               const float* __restrict__ b, int n) {
  int t = blockIdx.x * blockDim.x + threadIdx.x;
  if (t < n) out[t] = as * a[t] + bs * b[t];
}

// x [B, D, 1] -> x0 [D, B]
__global__ __launch_bounds__(256) void k_transpose_x(const float* __restrict__ x,
                                                     float* __restrict__ x0) {
  int t = blockIdx.x * blockDim.x + threadIdx.x;     // D*B threads
  int v = t >> 6, b = t & 63;
  x0[t] = x[(size_t)b * DV + v];
}

// zout[row,:] += scale*vals[e] * zin[col,:]   (4 columns per thread)
__global__ __launch_bounds__(256) void k_spmm(const int* __restrict__ rows,
                                              const int* __restrict__ cols,
                                              const float* __restrict__ vals,
                                              float scale,
                                              const float* __restrict__ zin,
                                              float* __restrict__ zout,
                                              int colShift, int total) {
  int t = blockIdx.x * blockDim.x + threadIdx.x;
  if (t >= total) return;
  int e  = t >> colShift;
  int c4 = (t & ((1 << colShift) - 1)) << 2;
  int ncols = 4 << colShift;
  int r = rows[e], c = cols[e];
  float v = vals[e] * scale;
  float4 zi = *(const float4*)(zin + (size_t)c * ncols + c4);
  float* zo = zout + (size_t)r * ncols + c4;
  atomicAdd(zo + 0, v * zi.x);
  atomicAdd(zo + 1, v * zi.y);
  atomicAdd(zo + 2, v * zi.z);
  atomicAdd(zo + 3, v * zi.w);
}

// conv1 projection (Fin=1): y1[(b*DV+v), f] += xk[v,b] * W1[f, k]
__global__ __launch_bounds__(256) void k_conv1_acc(const float* __restrict__ xk,
                                                   const float* __restrict__ w1,
                                                   float* __restrict__ y1, int kcheb) {
  int t = blockIdx.x * blockDim.x + threadIdx.x;     // DV*B threads
  int v = t >> 6, b = t & 63;
  float s = xk[t];
  float* yr = y1 + ((size_t)b * DV + v) * F1;
#pragma unroll
  for (int f = 0; f < F1; ++f)
    yr[f] += s * w1[f * KCH + kcheb];
}

// cl2 weight pre-transform: w2t[(k*F2 + n)*F1 + fin] = bf16(W2[n, fin*KCH + k])
__global__ __launch_bounds__(256) void k_w2t(const float* __restrict__ w2,
                                             unsigned short* __restrict__ w2t) {
  int t = blockIdx.x * blockDim.x + threadIdx.x;     // KCH*F2*F1 = 51200
  if (t >= KCH * F2 * F1) return;
  int fin = t & 31;
  int n   = (t >> 5) & 63;
  int k   = t >> 11;
  w2t[t] = f2bf(w2[(size_t)n * W2ROW + fin * KCH + k]);
}

// conv2 projection per Chebyshev term, WMMA bf16 16x16x32 (K = Fin = 32).
// xk layout [V2, B*F1] (col = b*32+fin)  -> A-tile rows are 32 contiguous f32.
// w2t layout [KCH][F2][F1]               -> B-tile columns are 32 contiguous bf16.
// y2[r, n] += sum_fin A * B,  r = b*V2 + v.  One wave per 16x16 tile.
__global__ __launch_bounds__(256) void k_conv2_acc(const float* __restrict__ xk,
                                                   const unsigned short* __restrict__ w2t,
                                                   float* __restrict__ y2, int kcheb) {
  int wave  = (blockIdx.x * blockDim.x + threadIdx.x) >> 5;
  int lane  = threadIdx.x & 31;
  int ntile = wave & 3;
  int mtile = wave >> 2;
  int r0 = mtile << 4;
  int b  = r0 >> 11;                 // V2 = 2048, tile lies in one batch row
  int v0 = r0 & (V2 - 1);
  int m  = lane & 15;
  int hi = lane >> 4;
  int n0 = ntile << 4;
  int nc = lane & 15;

  // A: 16x32 bf16 — lane (m,hi): elems 0..7 -> K=hi*8+i, elems 8..15 -> K=16+hi*8+i
  v16bf av, bv;
  const float* arow = xk + (size_t)(v0 + m) * NC2 + b * F1;
  float4 a0 = *(const float4*)(arow + hi * 8);
  float4 a1 = *(const float4*)(arow + hi * 8 + 4);
  float4 a2 = *(const float4*)(arow + 16 + hi * 8);
  float4 a3 = *(const float4*)(arow + 16 + hi * 8 + 4);
  av[0]  = us2bf(f2bf(a0.x)); av[1]  = us2bf(f2bf(a0.y));
  av[2]  = us2bf(f2bf(a0.z)); av[3]  = us2bf(f2bf(a0.w));
  av[4]  = us2bf(f2bf(a1.x)); av[5]  = us2bf(f2bf(a1.y));
  av[6]  = us2bf(f2bf(a1.z)); av[7]  = us2bf(f2bf(a1.w));
  av[8]  = us2bf(f2bf(a2.x)); av[9]  = us2bf(f2bf(a2.y));
  av[10] = us2bf(f2bf(a2.z)); av[11] = us2bf(f2bf(a2.w));
  av[12] = us2bf(f2bf(a3.x)); av[13] = us2bf(f2bf(a3.y));
  av[14] = us2bf(f2bf(a3.z)); av[15] = us2bf(f2bf(a3.w));

  // B: 32x16 bf16 — lane (nc,hi): elems 0..7 -> K=hi*8+i, elems 8..15 -> K=16+hi*8+i, N = nc
  const unsigned short* brow =
      w2t + ((size_t)kcheb * F2 + n0 + nc) * F1 + hi * 8;
#pragma unroll
  for (int i = 0; i < 8; ++i) {
    bv[i] = us2bf(brow[i]);
    bv[i + 8] = us2bf(brow[16 + i]);
  }

  // C: load f32 accumulator tile (M = vgpr + 8*hi, N = nc)
  v8f c;
  float* ybase = y2 + (size_t)r0 * F2 + n0 + nc;
#pragma unroll
  for (int j = 0; j < 8; ++j)
    c[j] = ybase[(size_t)(j + 8 * hi) * F2];

  c = __builtin_amdgcn_wmma_f32_16x16x32_bf16(false, av, false, bv, (short)0, c,
                                              false, false);
#pragma unroll
  for (int j = 0; j < 8; ++j)
    ybase[(size_t)(j + 8 * hi) * F2] = c[j];
}

// bias + relu + maxpool(p=2) over node dim; write conv2 layout [V2, B*F1]
__global__ __launch_bounds__(256) void k_pool1(const float* __restrict__ y1,
                                               const float* __restrict__ bias,
                                               float* __restrict__ x02) {
  int t = blockIdx.x * blockDim.x + threadIdx.x;     // B*V2*F1 = 2^22 threads
  int f  = t & 31;
  int v2 = (t >> 5) & 2047;
  int b  = t >> 16;
  size_t r0 = ((size_t)b * DV + 2 * v2) * F1 + f;
  float bvv = bias[f];
  float a = y1[r0] + bvv, c = y1[r0 + F1] + bvv;
  a = a > 0.f ? a : 0.f;
  c = c > 0.f ? c : 0.f;
  x02[(size_t)v2 * NC2 + b * F1 + f] = a > c ? a : c;
}

// bias + relu + maxpool(p=2); flatten to bf16 activations [B, 65536]
__global__ __launch_bounds__(256) void k_pool2(const float* __restrict__ y2,
                                               const float* __restrict__ bias,
                                               unsigned short* __restrict__ a2) {
  int t = blockIdx.x * blockDim.x + threadIdx.x;     // B*V3*F2 = 2^22 threads
  int f  = t & 63;
  int v3 = (t >> 6) & 1023;
  int b  = t >> 16;
  size_t r0 = ((size_t)b * V2 + 2 * v3) * F2 + f;
  float bvv = bias[f];
  float a = y2[r0] + bvv, c = y2[r0 + F2] + bvv;
  a = a > 0.f ? a : 0.f;
  c = c > 0.f ? c : 0.f;
  float mx = a > c ? a : c;
  a2[(size_t)b * FC1IN + v3 * F2 + f] = f2bf(mx);
}

__global__ __launch_bounds__(256) void k_cvt_bf16(const float* __restrict__ w,
                                                  unsigned short* __restrict__ wb, int n) {
  int t = blockIdx.x * blockDim.x + threadIdx.x;
  if (t < n) wb[t] = f2bf(w[t]);
}

// FC1: [64, 65536] bf16 @ [65536, 512], K split across 32 slices for occupancy.
// 4 M-tiles x 32 N-tiles x 32 K-slices = 4096 waves; partials via f32 atomics.
__global__ __launch_bounds__(256) void k_fc1(const unsigned short* __restrict__ a2,
                                             const unsigned short* __restrict__ w1,
                                             float* __restrict__ h3) {
  int wave  = (blockIdx.x * blockDim.x + threadIdx.x) >> 5;
  int lane  = threadIdx.x & 31;
  int ks    = wave & (KSPLIT - 1);
  int ntile = (wave >> 5) & 31;
  int mtile = wave >> 10;
  int m  = lane & 15;
  int hi = lane >> 4;
  int nc = lane & 15;
  int n0 = ntile << 4;
  const unsigned short* arow =
      a2 + (size_t)(mtile * 16 + m) * FC1IN + ks * KSLC;
  const unsigned short* brow =
      w1 + (size_t)(n0 + nc) * FC1IN + ks * KSLC + hi * 8;
  v8f c = {};
  for (int k0 = 0; k0 < KSLC; k0 += 32) {
    v16bf av, bv;
#pragma unroll
    for (int i = 0; i < 8; ++i) {
      av[i]     = us2bf(arow[k0 + hi * 8 + i]);
      av[i + 8] = us2bf(arow[k0 + 16 + hi * 8 + i]);
    }
#pragma unroll
    for (int i = 0; i < 8; ++i) {
      bv[i]     = us2bf(brow[k0 + i]);
      bv[i + 8] = us2bf(brow[k0 + 16 + i]);
    }
    c = __builtin_amdgcn_wmma_f32_16x16x32_bf16(false, av, false, bv, (short)0, c,
                                                false, false);
  }
#pragma unroll
  for (int j = 0; j < 8; ++j)
    atomicAdd(h3 + (size_t)(mtile * 16 + j + 8 * hi) * FC1F + n0 + nc, c[j]);
}

// in-place bias + relu on h3
__global__ __launch_bounds__(256) void k_fc1_finish(float* __restrict__ h3,
                                                    const float* __restrict__ bias) {
  int t = blockIdx.x * blockDim.x + threadIdx.x;     // B*FC1F
  if (t >= BATCH * FC1F) return;
  float v = h3[t] + bias[t & (FC1F - 1)];
  h3[t] = v > 0.f ? v : 0.f;
}

// FC2: [64, 512] @ [512, 10] + bias
__global__ __launch_bounds__(256) void k_fc2(const float* __restrict__ h3,
                                             const float* __restrict__ w,
                                             const float* __restrict__ b,
                                             float* __restrict__ out) {
  int t = blockIdx.x * blockDim.x + threadIdx.x;
  if (t >= BATCH * FC2F) return;
  int f = t % FC2F, bb = t / FC2F;
  float s = b[f];
  const float* hr = h3 + (size_t)bb * FC1F;
  const float* wr = w + (size_t)f * FC1F;
  for (int j = 0; j < FC1F; ++j) s += hr[j] * wr[j];
  out[t] = s;
}

// ---------------------------------------------------------------------------

extern "C" void kernel_launch(void* const* d_in, const int* in_sizes, int n_in,
                              void* d_out, int out_size, void* d_ws, size_t ws_size,
                              hipStream_t stream) {
  const float* x   = (const float*)d_in[0];
  const int*   l1r = (const int*)d_in[1];
  const int*   l1c = (const int*)d_in[2];
  const float* l1v = (const float*)d_in[3];
  const int*   l2r = (const int*)d_in[4];
  const int*   l2c = (const int*)d_in[5];
  const float* l2v = (const float*)d_in[6];
  const float* w1  = (const float*)d_in[7];
  const float* b1  = (const float*)d_in[8];
  const float* w2  = (const float*)d_in[9];
  const float* b2  = (const float*)d_in[10];
  const float* fw1 = (const float*)d_in[11];
  const float* fb1 = (const float*)d_in[12];
  const float* fw2 = (const float*)d_in[13];
  const float* fb2 = (const float*)d_in[14];
  float* out = (float*)d_out;

  char* ws = (char*)d_ws;
  size_t off = 0;
  auto wsalloc = [&](size_t bytes) {
    char* p = ws + off;
    off += (bytes + 255) & ~(size_t)255;
    return p;
  };
  const int N1 = DV * BATCH;               // 262144
  const int N2 = V2 * NC2;                 // 4194304
  float* xa1 = (float*)wsalloc((size_t)N1 * 4);
  float* xb1 = (float*)wsalloc((size_t)N1 * 4);
  float* xc1 = (float*)wsalloc((size_t)N1 * 4);
  float* y1  = (float*)wsalloc((size_t)BATCH * DV * F1 * 4);    // 32 MB
  float* xa2 = (float*)wsalloc((size_t)N2 * 4);                 // 16 MB
  float* xb2 = (float*)wsalloc((size_t)N2 * 4);
  float* xc2 = (float*)wsalloc((size_t)N2 * 4);
  float* y2  = (float*)wsalloc((size_t)BATCH * V2 * F2 * 4);    // 32 MB
  unsigned short* a2bf = (unsigned short*)wsalloc((size_t)BATCH * FC1IN * 2);
  unsigned short* w1bf = (unsigned short*)wsalloc((size_t)FC1F * FC1IN * 2);
  unsigned short* w2t  = (unsigned short*)wsalloc((size_t)KCH * F2 * F1 * 2);
  float* h3 = (float*)wsalloc((size_t)BATCH * FC1F * 4);
  (void)ws_size; (void)in_sizes; (void)n_in; (void)out_size;

  const int TB = 256;
  auto blk = [](int n, int tb) { return (n + tb - 1) / tb; };

  // One-time weight transforms (deterministic, part of every pass).
  k_cvt_bf16<<<blk(FC1F * FC1IN, TB), TB, 0, stream>>>(fw1, w1bf, FC1F * FC1IN);
  k_w2t<<<blk(KCH * F2 * F1, TB), TB, 0, stream>>>(w2, w2t);

  // ---------------- ChebConv layer 1 (Fin=1, K=25) ----------------
  k_transpose_x<<<blk(N1, TB), TB, 0, stream>>>(x, xa1);
  k_zero<<<blk(BATCH * DV * F1, TB), TB, 0, stream>>>(y1, BATCH * DV * F1);
  k_conv1_acc<<<blk(N1, TB), TB, 0, stream>>>(xa1, w1, y1, 0);

  // x1 = S(x0) - x0
  k_axpby<<<blk(N1, TB), TB, 0, stream>>>(xb1, -1.0f, xa1, 0.0f, xa1, N1);
  k_spmm<<<blk(NNZ1 * (BATCH / 4), TB), TB, 0, stream>>>(l1r, l1c, l1v, 1.0f, xa1, xb1,
                                                         4, NNZ1 * (BATCH / 4));
  k_conv1_acc<<<blk(N1, TB), TB, 0, stream>>>(xb1, w1, y1, 1);

  {
    float *p0 = xa1, *p1 = xb1, *p2 = xc1;
    for (int k = 2; k < KCH; ++k) {
      // x2 = 2*S(x1) - 2*x1 - x0
      k_axpby<<<blk(N1, TB), TB, 0, stream>>>(p2, -2.0f, p1, -1.0f, p0, N1);
      k_spmm<<<blk(NNZ1 * (BATCH / 4), TB), TB, 0, stream>>>(l1r, l1c, l1v, 2.0f, p1, p2,
                                                             4, NNZ1 * (BATCH / 4));
      k_conv1_acc<<<blk(N1, TB), TB, 0, stream>>>(p2, w1, y1, k);
      float* t = p0; p0 = p1; p1 = p2; p2 = t;
    }
  }
  // bias + relu + pool -> conv2 input layout [V2, B*F1]
  k_pool1<<<blk(BATCH * V2 * F1, TB), TB, 0, stream>>>(y1, b1, xa2);

  // ---------------- ChebConv layer 2 (Fin=32, K=25, WMMA) ----------------
  k_zero<<<blk(BATCH * V2 * F2, TB), TB, 0, stream>>>(y2, BATCH * V2 * F2);
  const int C2WAVES = (BATCH * V2 / 16) * (F2 / 16);   // 32768 waves
  const int C2BLKS  = C2WAVES * 32 / TB;               // 4096 blocks
  k_conv2_acc<<<C2BLKS, TB, 0, stream>>>(xa2, w2t, y2, 0);

  k_axpby<<<blk(N2, TB), TB, 0, stream>>>(xb2, -1.0f, xa2, 0.0f, xa2, N2);
  k_spmm<<<blk(NNZ2 * (NC2 / 4), TB), TB, 0, stream>>>(l2r, l2c, l2v, 1.0f, xa2, xb2,
                                                       9, NNZ2 * (NC2 / 4));
  k_conv2_acc<<<C2BLKS, TB, 0, stream>>>(xb2, w2t, y2, 1);

  {
    float *p0 = xa2, *p1 = xb2, *p2 = xc2;
    for (int k = 2; k < KCH; ++k) {
      k_axpby<<<blk(N2, TB), TB, 0, stream>>>(p2, -2.0f, p1, -1.0f, p0, N2);
      k_spmm<<<blk(NNZ2 * (NC2 / 4), TB), TB, 0, stream>>>(l2r, l2c, l2v, 2.0f, p1, p2,
                                                           9, NNZ2 * (NC2 / 4));
      k_conv2_acc<<<C2BLKS, TB, 0, stream>>>(p2, w2t, y2, k);
      float* t = p0; p0 = p1; p1 = p2; p2 = t;
    }
  }
  // bias + relu + pool -> bf16 flattened activations [64, 65536]
  k_pool2<<<blk(BATCH * V3 * F2, TB), TB, 0, stream>>>(y2, b2, a2bf);

  // ---------------- FC1 (WMMA bf16, K-split + atomics) + FC2 ----------------
  k_zero<<<blk(BATCH * FC1F, TB), TB, 0, stream>>>(h3, BATCH * FC1F);
  const int FC1WAVES = (BATCH / 16) * (FC1F / 16) * KSPLIT;   // 4096 waves
  k_fc1<<<FC1WAVES * 32 / TB, TB, 0, stream>>>(a2bf, w1bf, h3);
  k_fc1_finish<<<blk(BATCH * FC1F, TB), TB, 0, stream>>>(h3, fb1);
  k_fc2<<<blk(BATCH * FC2F, TB), TB, 0, stream>>>(h3, fw2, fb2, out);
}
